// Transformer_block_36043365548501
// MI455X (gfx1250) — compile-verified, matrix operand order checked by reading, not run on hardware
//
#include <hip/hip_runtime.h>

#define CCH   128
#define KNN   16
#define NHEAD 4
#define DHD   32
#define PTSW  16     // points per wave
#define PB    2      // points per GEMM batch sharing B fragments
#define NGRP  (PTSW / PB)
#define NW    2      // waves per block

typedef __attribute__((ext_vector_type(16))) _Float16 v16h;
typedef __attribute__((ext_vector_type(8)))  _Float16 v8h;
typedef __attribute__((ext_vector_type(8)))  float    v8f;

__device__ __forceinline__ float wsum32(float v) {
#pragma unroll
  for (int m = 16; m >= 1; m >>= 1) v += __shfl_xor(v, m, 32);
  return v;
}
__device__ __forceinline__ float hmax16(float v) {
#pragma unroll
  for (int m = 8; m >= 1; m >>= 1) v = fmaxf(v, __shfl_xor(v, m, 32));
  return v;
}
__device__ __forceinline__ float hsum16(float v) {
#pragma unroll
  for (int m = 8; m >= 1; m >>= 1) v += __shfl_xor(v, m, 32);
  return v;
}

__global__ __launch_bounds__(NW * 32, 1)
void pt_sa_block_kernel(const float* __restrict__ x,
                        const float* __restrict__ knn_feature,
                        const float* __restrict__ knn_xyz,
                        const float* __restrict__ q_w, const float* __restrict__ q_b,
                        const float* __restrict__ k_w, const float* __restrict__ k_b,
                        const float* __restrict__ v_w, const float* __restrict__ v_b,
                        const float* __restrict__ lin_w, const float* __restrict__ lin_b,
                        const float* __restrict__ ln1_g, const float* __restrict__ ln1_b,
                        const float* __restrict__ ln2_g, const float* __restrict__ ln2_b,
                        float* __restrict__ out, int n)
{
  // Block-shared weight fragment caches (f16, WMMA B layout): loaded once per block.
  __shared__ __align__(32) _Float16 KWf[8 * 5 * 32 * 16];       // 40 KB
  __shared__ __align__(32) _Float16 VWf[8 * 5 * 32 * 16];       // 40 KB
  // Per-wave working set
  __shared__ __align__(32) _Float16 NFY[NW][PB * 5 * 32 * 16];  // 10 KB/wave; reused as float Y
  __shared__ __align__(32) _Float16 Qh[NW][PTSW * CCH];         // 4 KB/wave
  __shared__ __align__(32) _Float16 KVT[NW][PB * CCH * KNN];    // 8 KB/wave, [col][row] transposed
  __shared__            float    X1s[NW][PTSW * CCH];           // 8 KB/wave
  __shared__            float    ATs[NW][PB * NHEAD * KNN];     // 0.5 KB/wave

  const int tid  = threadIdx.x;
  const int wv   = tid >> 5;
  const int lane = tid & 31;
  const int llo  = lane & 15;
  const int lhi  = lane >> 4;
  const long p0  = ((long)blockIdx.x * NW + wv) * PTSW;
  const float inv_d = 0.0883883476483f;   // 1/sqrt(128)
  const v8f vzero = {};

  _Float16* NF = NFY[wv];
  _Float16* Q  = Qh[wv];
  _Float16* KV = KVT[wv];
  float*    X1 = X1s[wv];
  float*    AT = ATs[wv];

  // ---------------- stage k_w / v_w as f16 B-fragments in LDS (once per block) ----------------
  {
    const float* W  = (wv == 0) ? k_w : v_w;
    _Float16*    WF = (wv == 0) ? KWf : VWf;
    for (int nt = 0; nt < 8; ++nt) {
#pragma unroll
      for (int t = 0; t < 5; ++t) {
        v16h fr;
#pragma unroll
        for (int j = 0; j < 16; ++j) {
          int kk  = (j >> 3) * 16 + lhi * 8 + (j & 7);
          int row = t * 32 + kk;
          fr[j] = (_Float16)((row < CCH + 3) ? W[row * CCH + nt * 16 + llo] : 0.f);
        }
        *(v16h*)&WF[((nt * 5 + t) * 32 + lane) * 16] = fr;
      }
    }
  }

  // ---------------- Q = x @ q_w + q_b  (M=16 points, K=128, N=128) ----------------
  {
    long prow = p0 + llo; if (prow >= n) prow = n - 1;
    v16h afr[4];
#pragma unroll
    for (int t = 0; t < 4; ++t) {
#pragma unroll
      for (int j = 0; j < 16; ++j) {
        int kk = (j >> 3) * 16 + lhi * 8 + (j & 7);
        afr[t][j] = (_Float16)x[prow * CCH + t * 32 + kk];
      }
    }
    for (int nt = 0; nt < 8; ++nt) {
      v8f acc = vzero;
#pragma unroll
      for (int t = 0; t < 4; ++t) {
        v16h bfr;
#pragma unroll
        for (int j = 0; j < 16; ++j) {
          int kk = (j >> 3) * 16 + lhi * 8 + (j & 7);
          bfr[j] = (_Float16)q_w[(t * 32 + kk) * CCH + nt * 16 + llo];
        }
        acc = __builtin_amdgcn_wmma_f32_16x16x32_f16(false, afr[t], false, bfr,
                                                     (short)0, acc, false, false);
      }
      int col = nt * 16 + llo;
      float bias = q_b[col];
#pragma unroll
      for (int r = 0; r < 8; ++r)
        Q[(r + lhi * 8) * CCH + col] = (_Float16)(acc[r] + bias);
    }
  }
  __syncthreads();   // weight cache + Q visible

  // ---------------- per-batch attention ----------------
  for (int g = 0; g < NGRP; ++g) {
    if (g + 1 < NGRP) {   // prefetch next batch's neighbor features
      long pn = p0 + (long)(g + 1) * PB; if (pn >= n) pn = n - 1;
      __builtin_prefetch(&knn_feature[(pn * KNN + llo) * CCH], 0, 1);
    }

    // stage nf = [knn_feature | knn_xyz | pad] A-fragments (f16 WMMA layout, own-lane slots)
    for (int pp = 0; pp < PB; ++pp) {
      long pg = p0 + g * PB + pp; if (pg >= n) pg = n - 1;
#pragma unroll
      for (int t = 0; t < 5; ++t) {
        v16h fr;
#pragma unroll
        for (int j = 0; j < 16; ++j) {
          int kk = (j >> 3) * 16 + lhi * 8 + (j & 7);
          int cc = t * 32 + kk;
          float v = 0.f;
          if (cc < CCH)          v = knn_feature[(pg * KNN + llo) * CCH + cc];
          else if (cc < CCH + 3) v = knn_xyz[(pg * KNN + llo) * 3 + (cc - CCH)];
          fr[j] = (_Float16)v;
        }
        *(v16h*)&NF[((pp * 5 + t) * 32 + lane) * 16] = fr;
      }
    }
    __syncthreads();

    // ---- Kt = nf @ k_w + k_b : pure LDS->WMMA inner loop ----
    for (int nt = 0; nt < 8; ++nt) {
      v8f acc[PB];
#pragma unroll
      for (int pp = 0; pp < PB; ++pp) acc[pp] = vzero;
#pragma unroll
      for (int t = 0; t < 5; ++t) {
        v16h bfr = *(const v16h*)&KWf[((nt * 5 + t) * 32 + lane) * 16];
#pragma unroll
        for (int pp = 0; pp < PB; ++pp) {
          v16h a = *(const v16h*)&NF[((pp * 5 + t) * 32 + lane) * 16];
          acc[pp] = __builtin_amdgcn_wmma_f32_16x16x32_f16(false, a, false, bfr,
                                                           (short)0, acc[pp], false, false);
        }
      }
      int col = nt * 16 + llo;
      float bias = k_b[col];
#pragma unroll
      for (int pp = 0; pp < PB; ++pp) {
        v8h hv;
#pragma unroll
        for (int r = 0; r < 8; ++r) hv[r] = (_Float16)(acc[pp][r] + bias);
        *(v8h*)&KV[(pp * CCH + col) * KNN + lhi * 8] = hv;   // transposed [col][row]
      }
    }
    __syncthreads();

    // ---- attention scores + softmax: lane owns (h=lhi, k=llo) and (h+2, k) ----
    for (int pp = 0; pp < PB; ++pp) {
      int pi = g * PB + pp;
      int h1 = lhi, k1 = llo, h2 = h1 + 2;
      int r1 = h1 * 4 + (k1 >> 2), r2 = h2 * 4 + (k1 >> 2);
      int cb = (k1 & 3) * DHD;
      const _Float16* q1 = &Q[pi * CCH + h1 * DHD];
      const _Float16* q2 = &Q[pi * CCH + h2 * DHD];
      float a1 = 0.f, a2 = 0.f;
#pragma unroll
      for (int d = 0; d < DHD; ++d) {
        a1 += (float)q1[d] * (float)KV[(pp * CCH + cb + d) * KNN + r1];
        a2 += (float)q2[d] * (float)KV[(pp * CCH + cb + d) * KNN + r2];
      }
      a1 *= inv_d; a2 *= inv_d;
      float m1 = hmax16(a1), m2 = hmax16(a2);
      float e1 = __expf(a1 - m1), e2 = __expf(a2 - m2);
      float s1 = hsum16(e1), s2 = hsum16(e2);
      AT[pp * 64 + lane]      = e1 / s1;
      AT[pp * 64 + lane + 32] = e2 / s2;
    }
    __syncthreads();

    // ---- Vt = nf @ v_w + v_b (overwrites KV) ----
    for (int nt = 0; nt < 8; ++nt) {
      v8f acc[PB];
#pragma unroll
      for (int pp = 0; pp < PB; ++pp) acc[pp] = vzero;
#pragma unroll
      for (int t = 0; t < 5; ++t) {
        v16h bfr = *(const v16h*)&VWf[((nt * 5 + t) * 32 + lane) * 16];
#pragma unroll
        for (int pp = 0; pp < PB; ++pp) {
          v16h a = *(const v16h*)&NF[((pp * 5 + t) * 32 + lane) * 16];
          acc[pp] = __builtin_amdgcn_wmma_f32_16x16x32_f16(false, a, false, bfr,
                                                           (short)0, acc[pp], false, false);
        }
      }
      int col = nt * 16 + llo;
      float bias = v_b[col];
#pragma unroll
      for (int pp = 0; pp < PB; ++pp) {
        v8h hv;
#pragma unroll
        for (int r = 0; r < 8; ++r) hv[r] = (_Float16)(acc[pp][r] + bias);
        *(v8h*)&KV[(pp * CCH + col) * KNN + lhi * 8] = hv;
      }
    }
    __syncthreads();

    // ---- att_feat + residual + LN1; lane owns channels c = i*32 + lane ----
    for (int pp = 0; pp < PB; ++pp) {
      long pg = p0 + g * PB + pp; if (pg >= n) pg = n - 1;
      int pi = g * PB + pp;
      float sv[4];
#pragma unroll
      for (int i = 0; i < 4; ++i) {
        float af = 0.f;
#pragma unroll
        for (int k = 0; k < KNN; ++k)
          af += AT[pp * 64 + i * KNN + k] *
                (float)KV[(pp * CCH + (k & 3) * DHD + lane) * KNN + (i * 4 + (k >> 2))];
        sv[i] = x[pg * CCH + (i * DHD + lane)] + af;
      }
      float mean = wsum32(sv[0] + sv[1] + sv[2] + sv[3]) * (1.f / CCH);
      float var = 0.f;
#pragma unroll
      for (int i = 0; i < 4; ++i) { float dd = sv[i] - mean; var += dd * dd; }
      var = wsum32(var) * (1.f / CCH);
      float rs = rsqrtf(var + 1e-5f);
#pragma unroll
      for (int i = 0; i < 4; ++i) {
        int c = i * DHD + lane;
        X1[pi * CCH + c] = (sv[i] - mean) * rs * ln1_g[c] + ln1_b[c];
      }
    }
    __syncthreads();
  }

  // ---------------- y = x1 @ lin_w  (M=16 points); Y aliases dead NF region ----------------
  float* Y = (float*)NF;
  {
    v16h afr[4];
#pragma unroll
    for (int t = 0; t < 4; ++t) {
#pragma unroll
      for (int j = 0; j < 16; ++j) {
        int kk = (j >> 3) * 16 + lhi * 8 + (j & 7);
        afr[t][j] = (_Float16)X1[llo * CCH + t * 32 + kk];
      }
    }
    for (int nt = 0; nt < 8; ++nt) {
      v8f acc = vzero;
#pragma unroll
      for (int t = 0; t < 4; ++t) {
        v16h bfr;
#pragma unroll
        for (int j = 0; j < 16; ++j) {
          int kk = (j >> 3) * 16 + lhi * 8 + (j & 7);
          bfr[j] = (_Float16)lin_w[(t * 32 + kk) * CCH + nt * 16 + llo];
        }
        acc = __builtin_amdgcn_wmma_f32_16x16x32_f16(false, afr[t], false, bfr,
                                                     (short)0, acc, false, false);
      }
      int col = nt * 16 + llo;
#pragma unroll
      for (int r = 0; r < 8; ++r)
        Y[(r + lhi * 8) * CCH + col] = acc[r];
    }
  }
  __syncthreads();

  // ---------------- x2 = LN2(x1 + y + lin_b) ----------------
  for (int pi = 0; pi < PTSW; ++pi) {
    long pg = p0 + pi;
    bool valid = pg < n;
    float sv[4];
#pragma unroll
    for (int i = 0; i < 4; ++i) {
      int c = i * DHD + lane;
      sv[i] = X1[pi * CCH + c] + Y[pi * CCH + c] + lin_b[c];
    }
    float mean = wsum32(sv[0] + sv[1] + sv[2] + sv[3]) * (1.f / CCH);
    float var = 0.f;
#pragma unroll
    for (int i = 0; i < 4; ++i) { float dd = sv[i] - mean; var += dd * dd; }
    var = wsum32(var) * (1.f / CCH);
    float rs = rsqrtf(var + 1e-5f);
    if (valid) {
#pragma unroll
      for (int i = 0; i < 4; ++i) {
        int c = i * DHD + lane;
        out[pg * CCH + c] = (sv[i] - mean) * rs * ln2_g[c] + ln2_b[c];
      }
    }
  }
}

extern "C" void kernel_launch(void* const* d_in, const int* in_sizes, int n_in,
                              void* d_out, int out_size, void* d_ws, size_t ws_size,
                              hipStream_t stream) {
  (void)n_in; (void)d_ws; (void)ws_size; (void)out_size;
  const float* x    = (const float*)d_in[0];
  const float* knnf = (const float*)d_in[1];
  const float* knnx = (const float*)d_in[2];
  const float* q_w  = (const float*)d_in[3];
  const float* q_b  = (const float*)d_in[4];
  const float* k_w  = (const float*)d_in[5];
  const float* k_b  = (const float*)d_in[6];
  const float* v_w  = (const float*)d_in[7];
  const float* v_b  = (const float*)d_in[8];
  const float* l_w  = (const float*)d_in[9];
  const float* l_b  = (const float*)d_in[10];
  const float* g1   = (const float*)d_in[11];
  const float* b1   = (const float*)d_in[12];
  const float* g2   = (const float*)d_in[13];
  const float* b2   = (const float*)d_in[14];
  float* out = (float*)d_out;
  int n = in_sizes[0] / CCH;
  int pts_per_block = NW * PTSW;
  int blocks = (n + pts_per_block - 1) / pts_per_block;
  hipLaunchKernelGGL(pt_sa_block_kernel, dim3(blocks), dim3(NW * 32), 0, stream,
                     x, knnf, knnx, q_w, q_b, k_w, k_b, v_w, v_b,
                     l_w, l_b, g1, b1, g2, b2, out, n);
}
